// UnitWiseGRU_49744311222542
// MI455X (gfx1250) — matrix operand order repaired and checked by reading, not compiled, not run on hardware
//
#include <hip/hip_runtime.h>

// UnitWiseGRU for MI455X (gfx1250): per-unit 64x64 GEMMs via v_wmma_f32_16x16x32_bf16.
// One block per unit; weights converted to bf16 and pre-swizzled into WMMA B-fragment
// layout in LDS; x/h streamed once per row-tile as A fragments; f32 accumulate + f32
// sigmoid/tanh epilogue. Memory-bound design (~240 MB total traffic, ~10-13 us floor
// at 23.3 TB/s).
//
// B fragments are re-read from LDS every iteration through *volatile address_space(3)*
// loads: volatile defeats LICM (round 1 hoisted all 48 fragments and spilled to
// scratch), and the explicit LDS address space keeps the loads as ds_load_b128 with
// immediate offsets (round 2's volatile generic loads became serialized SCOPE_SYS
// flat loads with 64-bit address registers).

typedef __attribute__((ext_vector_type(16))) __bf16        v16bf;
typedef __attribute__((ext_vector_type(8)))  float         v8f;
typedef __attribute__((ext_vector_type(4)))  float         f4;
typedef __attribute__((ext_vector_type(4)))  unsigned int  v4u;

typedef __attribute__((address_space(3))) const volatile v4u* lds_v4u_p;

#define N_UNITS 512
#define DIM     64
#define BATCH   512

__device__ __forceinline__ v8f wmma_bf16(v16bf a, v16bf b, v8f c) {
  // (neg_a, A, neg_b, B, c_mod, C, reuse_a, reuse_b)
  return __builtin_amdgcn_wmma_f32_16x16x32_bf16(false, a, false, b, (short)0, c, false, false);
}

// Non-hoistable LDS fragment read: two volatile LDS-space 16B loads -> ds_load_b128 x2.
// q must point at the lane's 32B fragment row; offsets fold into ds immediates.
__device__ __forceinline__ v16bf lds_frag(lds_v4u_p q) {
  union { v4u u[2]; v16bf b; } f;
  f.u[0] = q[0];
  f.u[1] = q[1];
  return f.b;
}

__device__ __forceinline__ float fast_sigmoid(float v) {
  return __builtin_amdgcn_rcpf(1.0f + __expf(-v));
}

// Build a 16x32 bf16 A-fragment for this lane from a row of 64 f32 values.
// ISA layout: lane<16 holds K = kb..kb+7 and kb+16..kb+23 (kb = 0 for lanes 0-15,
// kb = 8 for lanes 16-31), all for row M = lane&15.  kbase = kt*32 + kb.
__device__ __forceinline__ v16bf load_a_frag(const float* __restrict__ rowp, int kbase) {
  f4 a0 = *(const f4*)(rowp + kbase);
  f4 a1 = *(const f4*)(rowp + kbase + 4);
  f4 a2 = *(const f4*)(rowp + kbase + 16);
  f4 a3 = *(const f4*)(rowp + kbase + 20);
  v16bf r;
  r[0]  = (__bf16)a0.x; r[1]  = (__bf16)a0.y; r[2]  = (__bf16)a0.z; r[3]  = (__bf16)a0.w;
  r[4]  = (__bf16)a1.x; r[5]  = (__bf16)a1.y; r[6]  = (__bf16)a1.z; r[7]  = (__bf16)a1.w;
  r[8]  = (__bf16)a2.x; r[9]  = (__bf16)a2.y; r[10] = (__bf16)a2.z; r[11] = (__bf16)a2.w;
  r[12] = (__bf16)a3.x; r[13] = (__bf16)a3.y; r[14] = (__bf16)a3.z; r[15] = (__bf16)a3.w;
  return r;
}

__global__ __launch_bounds__(256) void UnitWiseGRU_kernel(
    const float* __restrict__ x,   const float* __restrict__ h,
    const float* __restrict__ Wir, const float* __restrict__ bir,
    const float* __restrict__ Whr, const float* __restrict__ bhr,
    const float* __restrict__ Wiz, const float* __restrict__ biz,
    const float* __restrict__ Whz, const float* __restrict__ bhz,
    const float* __restrict__ Win, const float* __restrict__ bin_,
    const float* __restrict__ Whn, const float* __restrict__ bhn,
    float* __restrict__ out)
{
  // Weights in B-fragment order: [matrix][fragId = kt*4+nt][lane][16 bf16 slots]
  __shared__ alignas(32) __bf16 wsw[6][8][32][16];   // 48 KB
  __shared__ float sbias[6][DIM];                     // 1.5 KB

  const int u   = blockIdx.x;
  const int tid = threadIdx.x;

  const float* Ws[6] = {Wir, Whr, Wiz, Whz, Win, Whn};
  const float* Bs[6] = {bir, bhr, biz, bhz, bin_, bhn};

  // Phase 1: load + convert + swizzle this unit's weights/biases into LDS.
  for (int m = 0; m < 6; ++m) {
    const float* W = Ws[m] + (size_t)u * DIM * DIM;   // row-major [k][n]
    for (int idx = tid; idx < DIM * DIM; idx += 256) {
      int k   = idx >> 6, n = idx & 63;
      int kt  = k >> 5,  nt = n >> 4;
      int kin = k & 31,  nin = n & 15;
      int ln  = nin + ((kin & 8) ? 16 : 0);           // B layout: lanes16-31 hold K 8-15,24-31
      wsw[m][kt * 4 + nt][ln][(kin & 7) + ((kin >> 4) << 3)] = (__bf16)W[idx];
    }
    if (tid < DIM) sbias[m][tid] = Bs[m][(size_t)u * DIM + tid];
  }
  __syncthreads();

  const int lane = tid & 31;
  const int wv   = tid >> 5;       // wave id 0..7
  const int half = lane >> 4;      // 0 / 1
  const int lidx = lane & 15;
  const int kb   = half * 8;       // A-fragment K sub-base for this lane

  // LDS base pointer for this lane's fragment rows (AS3, 32-bit addressing).
  lds_v4u_p wbase = (lds_v4u_p)&wsw[0][0][lane][0];

  #pragma unroll 1
  for (int bt = 0; bt < BATCH; bt += 128) {
    const int row_base = bt + wv * 16;
    const int arow     = row_base + lidx;
    const float* xrow  = x + ((size_t)arow * N_UNITS + u) * DIM;
    const float* hrow  = h + ((size_t)arow * N_UNITS + u) * DIM;

    v16bf xa0 = load_a_frag(xrow, kb);        // K 0..31
    v16bf xa1 = load_a_frag(xrow, 32 + kb);   // K 32..63
    v16bf ha0 = load_a_frag(hrow, kb);
    v16bf ha1 = load_a_frag(hrow, 32 + kb);

    #pragma unroll
    for (int nt = 0; nt < 4; ++nt) {
      const int ncol = nt * 16 + lidx;        // global output column for this lane
      const float brz = sbias[0][ncol] + sbias[1][ncol];
      const float bzz = sbias[2][ncol] + sbias[3][ncol];
      const float bgi = sbias[4][ncol];
      const float bgh = sbias[5][ncol];
      v8f accr, accz, acci, acch;
      #pragma unroll
      for (int i = 0; i < 8; ++i) { accr[i] = brz; accz[i] = bzz; acci[i] = bgi; acch[i] = bgh; }

      // Fragment (m, kt, nt) lives at v4u offset ((m*8 + kt*4 + nt) * 64) from wbase
      // (each fragment = 32 lanes * 32 B = 64 v4u; lane offset is inside wbase).
      #define BFRAG(mm, kt) lds_frag(wbase + (((mm) * 8 + (kt) * 4 + nt) * 64))
      // r gate: x@W_ir + h@W_hr (+ biases already in C)
      accr = wmma_bf16(xa0, BFRAG(0, 0), accr);
      accr = wmma_bf16(xa1, BFRAG(0, 1), accr);
      accr = wmma_bf16(ha0, BFRAG(1, 0), accr);
      accr = wmma_bf16(ha1, BFRAG(1, 1), accr);
      // z gate: x@W_iz + h@W_hz
      accz = wmma_bf16(xa0, BFRAG(2, 0), accz);
      accz = wmma_bf16(xa1, BFRAG(2, 1), accz);
      accz = wmma_bf16(ha0, BFRAG(3, 0), accz);
      accz = wmma_bf16(ha1, BFRAG(3, 1), accz);
      // n gate pieces: gi = x@W_in + b_in ; gh = h@W_hn + b_hn (kept separate for r*gh)
      acci = wmma_bf16(xa0, BFRAG(4, 0), acci);
      acci = wmma_bf16(xa1, BFRAG(4, 1), acci);
      acch = wmma_bf16(ha0, BFRAG(5, 0), acch);
      acch = wmma_bf16(ha1, BFRAG(5, 1), acch);
      #undef BFRAG

      // Epilogue: C/D layout m = vgpr + 8*half, n = lane&15.
      #pragma unroll
      for (int v = 0; v < 8; ++v) {
        const int    m = v + half * 8;
        const size_t o = ((size_t)(row_base + m) * N_UNITS + u) * DIM + ncol;
        const float hv = h[o];
        const float rr = fast_sigmoid(accr[v]);
        const float zz = fast_sigmoid(accz[v]);
        const float ng = acci[v] + rr * acch[v];
        const float nn = 2.0f * fast_sigmoid(2.0f * ng) - 1.0f;   // tanh
        out[o] = (1.0f - zz) * nn + zz * hv;
      }
    }
  }
}

extern "C" void kernel_launch(void* const* d_in, const int* in_sizes, int n_in,
                              void* d_out, int out_size, void* d_ws, size_t ws_size,
                              hipStream_t stream) {
  (void)in_sizes; (void)n_in; (void)out_size; (void)d_ws; (void)ws_size;
  const float* x    = (const float*)d_in[0];
  const float* h    = (const float*)d_in[1];
  const float* Wir  = (const float*)d_in[2];
  const float* bir  = (const float*)d_in[3];
  const float* Whr  = (const float*)d_in[4];
  const float* bhr  = (const float*)d_in[5];
  const float* Wiz  = (const float*)d_in[6];
  const float* biz  = (const float*)d_in[7];
  const float* Whz  = (const float*)d_in[8];
  const float* bhz  = (const float*)d_in[9];
  const float* Win  = (const float*)d_in[10];
  const float* bin_ = (const float*)d_in[11];
  const float* Whn  = (const float*)d_in[12];
  const float* bhn  = (const float*)d_in[13];
  float* out = (float*)d_out;

  UnitWiseGRU_kernel<<<dim3(N_UNITS), dim3(256), 0, stream>>>(
      x, h, Wir, bir, Whr, bhr, Wiz, biz, Whz, bhz, Win, bin_, Whn, bhn, out);
}
